// UnstructuredNetwork_73650099192099
// MI455X (gfx1250) — matrix operand order repaired and checked by reading, not compiled, not run on hardware
//
#include <hip/hip_runtime.h>

typedef __attribute__((ext_vector_type(16))) _Float16 v16h;
typedef __attribute__((ext_vector_type(8)))  float    v8f;
typedef __attribute__((ext_vector_type(4)))  float    f4;

#define HID     20
#define NLAYERS 50
#define DIN     3
#define DOUT    3

// ---- workspace layout (bytes) ----
// A-fragments (f16): 100 matmuls x 2 tiles x 32 lanes x 16 halfs
#define NFRAG_HALFS (100 * 2 * 32 * 16)          // 102400 halfs = 204800 B
#define OFF_BIAS    (NFRAG_HALFS * 2)            // padded biases: 100 x 32 f32
#define OFF_LIFT    (OFF_BIAS + 100 * 32 * 4)    // lift: 32 x f4 {w0,w1,w2,b}
#define OFF_PROJ    (OFF_LIFT + 32 * 16)         // proj: 32 x f4 {p0,p1,p2,0}

// Hidden-feature slot permutation: B-fragment slot k carries feature sigma(k).
// Chosen so that the D/C accumulator layout (features 8-per-lane-group) maps
// lane-locally onto the B layout (K 16-per-lane-group): no cross-lane moves.
__host__ __device__ __forceinline__ int sigma_slot(int k) {
    int gg = k >> 4, jj = k & 15;
    return (jj < 8) ? (8 * gg + jj) : (16 + 8 * gg + (jj - 8));
}

// ---------------------------------------------------------------------------
// Prep kernel: reshape weights into per-lane WMMA A-fragments (f16, padded
// 20->32, columns permuted by sigma), padded biases, lift/proj tables.
// A register layout (16x32 f16, ISA 7.12.2): lane l -> row m = tile*16+(l%16),
// g = l>>4; half j<8 -> kslot = 8g+j ; half j>=8 -> kslot = 16 + 8g + (j-8).
// Matrix content: A[m][kslot] = W[kslot][m]   (i.e. W^T; kslot = sigma(16g+j))
// ---------------------------------------------------------------------------
__global__ void prep_kernel(const float* __restrict__ lift_w,
                            const float* __restrict__ lift_b,
                            const float* __restrict__ W1,
                            const float* __restrict__ b1,
                            const float* __restrict__ W2,
                            const float* __restrict__ b2,
                            const float* __restrict__ proj_w,
                            char* __restrict__ ws) {
    int idx = blockIdx.x * blockDim.x + threadIdx.x;
    _Float16* frag  = (_Float16*)ws;
    float*    bpad  = (float*)(ws + OFF_BIAS);
    f4*       liftp = (f4*)(ws + OFF_LIFT);
    f4*       projp = (f4*)(ws + OFF_PROJ);

    if (idx < 6400) {                       // one thread per (matmul,tile,lane)
        int l     = idx & 31;
        int t     = (idx >> 5) & 1;
        int m     = idx >> 6;               // matmul 0..99
        int layer = m >> 1;
        int which = m & 1;                  // 0 -> W1, 1 -> W2
        const float* W = which ? W2 : W1;
        int g    = l >> 4;
        int mrow = t * 16 + (l & 15);       // output feature (row of W^T)
        for (int j = 0; j < 16; ++j) {
            int kslot = (j < 8) ? (8 * g + j) : (16 + 8 * g + (j - 8));
            int kfeat = kslot;              // logical input feature
            float v = 0.0f;
            if (kfeat < HID && mrow < HID)
                v = W[(layer * HID + kfeat) * HID + mrow];
            frag[idx * 16 + j] = (_Float16)v;
        }
    } else if (idx < 6400 + 3200) {         // padded biases
        int i2 = idx - 6400;
        int m  = i2 >> 5;
        int i  = i2 & 31;
        int layer = m >> 1;
        int which = m & 1;
        float v = 0.0f;
        if (i < HID) v = (which ? b2 : b1)[layer * HID + i];
        bpad[m * 32 + i] = v;
    } else if (idx < 6400 + 3200 + 32) {    // lift table
        int f = idx - 6400 - 3200;
        f4 v = {0.f, 0.f, 0.f, 0.f};
        if (f < HID) {
            v.x = lift_w[0 * HID + f];
            v.y = lift_w[1 * HID + f];
            v.z = lift_w[2 * HID + f];
            v.w = lift_b[f];
        }
        liftp[f] = v;
    } else if (idx < 6400 + 3200 + 64) {    // proj table
        int f = idx - 6400 - 3200 - 32;
        f4 v = {0.f, 0.f, 0.f, 0.f};
        if (f < HID) {
            v.x = proj_w[f * DOUT + 0];
            v.y = proj_w[f * DOUT + 1];
            v.z = proj_w[f * DOUT + 2];
        }
        projp[f] = v;
    }
}

__device__ __forceinline__ float leaky(float z) { return fmaxf(z, 0.01f * z); }

// Build a B fragment (32x16 f16) from f32 activations in D/C layout.
// With the sigma column permutation this is purely lane-local packing.
__device__ __forceinline__ v16h build_b(const v8f& t0, const v8f& t1) {
    v16h b;
#pragma unroll
    for (int j = 0; j < 8; ++j) {
        b[j]     = (_Float16)t0[j];
        b[j + 8] = (_Float16)t1[j];
    }
    return b;
}

// Same, with leaky applied in packed f16 after conversion (values feed a f16
// WMMA B operand anyway): v_cvt_pk + v_pk_mul_f16 + v_pk_max_num_f16.
__device__ __forceinline__ v16h build_b_act(const v8f& t0, const v8f& t1) {
    v16h b = build_b(t0, t1);
    v16h s;
#pragma unroll
    for (int j = 0; j < 16; ++j) s[j] = (_Float16)0.01f;  // folds to splat
    s = b * s;
    return __builtin_elementwise_max(b, s);
}

#define WMMA16(A, Bv, Cv) \
    __builtin_amdgcn_wmma_f32_16x16x32_f16(false, (A), false, (Bv), (short)0, (Cv), false, false)

// ---------------------------------------------------------------------------
// Main kernel: one wave processes 32 points (two independent 16-point groups
// u, v) through all 50 layers. The two groups share A/C fragments; their
// interleaved, mutually independent WMMAs fill the WMMA->VALU hazard slots
// and halve weight vmem per point. Activations live in registers (f32, WMMA
// C/D layout); all matmuls are v_wmma_f32_16x16x32_f16 with bias as C.
// ---------------------------------------------------------------------------
__global__ __launch_bounds__(256) void mlp_kernel(
    const float* __restrict__ x, const float* __restrict__ dts,
    const float* __restrict__ proj_b, const char* __restrict__ ws,
    float* __restrict__ out, int B) {
    const int lane = threadIdx.x & 31;
    const int wave = threadIdx.x >> 5;
    const int tile = blockIdx.x * 8 + wave;
    const int n    = lane & 15;
    const int g    = lane >> 4;

    int pu = tile * 32 + n;                  // group u point
    int pv = tile * 32 + 16 + n;             // group v point
    const bool vu = (pu < B), vv = (pv < B);
    if (pu >= B) pu = B - 1;
    if (pv >= B) pv = B - 1;

    const _Float16* frag  = (const _Float16*)ws;
    const float*    bpad  = (const float*)(ws + OFF_BIAS);
    const f4*       liftp = (const f4*)(ws + OFF_LIFT);
    const f4*       projp = (const f4*)(ws + OFF_PROJ);

    const float xu0 = x[pu * 3 + 0], xu1 = x[pu * 3 + 1], xu2 = x[pu * 3 + 2];
    const float xv0 = x[pv * 3 + 0], xv1 = x[pv * 3 + 1], xv2 = x[pv * 3 + 2];

    // lift into D-layout registers: *0[r] = feat 8g+r, *1[r] = feat 16+8g+r
    v8f hu0, hu1, hv0, hv1;
#pragma unroll
    for (int r = 0; r < 8; ++r) {
        f4 a = liftp[8 * g + r];
        hu0[r] = a.w + xu0 * a.x + xu1 * a.y + xu2 * a.z;
        hv0[r] = a.w + xv0 * a.x + xv1 * a.y + xv2 * a.z;
        f4 b = liftp[16 + 8 * g + r];
        hu1[r] = b.w + xu0 * b.x + xu1 * b.y + xu2 * b.z;
        hv1[r] = b.w + xv0 * b.x + xv1 * b.y + xv2 * b.z;
    }

    for (int layer = 0; layer < NLAYERS; ++layer) {
        const int m1 = 2 * layer, m2 = 2 * layer + 1;

        v16h bu = build_b(hu0, hu1);
        v16h bv = build_b(hv0, hv1);
        v16h a0 = *(const v16h*)(frag + ((m1 * 2 + 0) * 32 + lane) * 16);
        v16h a1 = *(const v16h*)(frag + ((m1 * 2 + 1) * 32 + lane) * 16);
        v8f  c0 = *(const v8f*)(bpad + m1 * 32 + 0 * 16 + 8 * g);
        v8f  c1 = *(const v8f*)(bpad + m1 * 32 + 1 * 16 + 8 * g);
        v8f zu0 = WMMA16(a0, bu, c0);
        v8f zv0 = WMMA16(a0, bv, c0);
        v8f zu1 = WMMA16(a1, bu, c1);
        v8f zv1 = WMMA16(a1, bv, c1);

        v16h bu2 = build_b_act(zu0, zu1);    // leaky in packed f16
        v16h bv2 = build_b_act(zv0, zv1);
        v16h a2 = *(const v16h*)(frag + ((m2 * 2 + 0) * 32 + lane) * 16);
        v16h a3 = *(const v16h*)(frag + ((m2 * 2 + 1) * 32 + lane) * 16);
        v8f  c2 = *(const v8f*)(bpad + m2 * 32 + 0 * 16 + 8 * g);
        v8f  c3 = *(const v8f*)(bpad + m2 * 32 + 1 * 16 + 8 * g);
        v8f yu0 = WMMA16(a2, bu2, c2);
        v8f yv0 = WMMA16(a2, bv2, c2);
        v8f yu1 = WMMA16(a3, bu2, c3);
        v8f yv1 = WMMA16(a3, bv2, c3);

        const float dt = dts[layer];
#pragma unroll
        for (int j = 0; j < 8; ++j) {        // f32 residual update
            hu0[j] += dt * leaky(yu0[j]);
            hu1[j] += dt * leaky(yu1[j]);
            hv0[j] += dt * leaky(yv0[j]);
            hv1[j] += dt * leaky(yv1[j]);
        }
    }

    // projection 20->3: per-lane partials over this lane's 16 features,
    // then half-wave reduction (lanes l and l^16 hold the same point).
    float u0 = 0.f, u1 = 0.f, u2 = 0.f;
    float w0 = 0.f, w1 = 0.f, w2 = 0.f;
#pragma unroll
    for (int r = 0; r < 8; ++r) {
        f4 pa = projp[8 * g + r];
        u0 += hu0[r] * pa.x; u1 += hu0[r] * pa.y; u2 += hu0[r] * pa.z;
        w0 += hv0[r] * pa.x; w1 += hv0[r] * pa.y; w2 += hv0[r] * pa.z;
        f4 pb = projp[16 + 8 * g + r];
        u0 += hu1[r] * pb.x; u1 += hu1[r] * pb.y; u2 += hu1[r] * pb.z;
        w0 += hv1[r] * pb.x; w1 += hv1[r] * pb.y; w2 += hv1[r] * pb.z;
    }
    u0 += __shfl_xor(u0, 16, 32);
    u1 += __shfl_xor(u1, 16, 32);
    u2 += __shfl_xor(u2, 16, 32);
    w0 += __shfl_xor(w0, 16, 32);
    w1 += __shfl_xor(w1, 16, 32);
    w2 += __shfl_xor(w2, 16, 32);
    const float pb0 = proj_b[0], pb1 = proj_b[1], pb2 = proj_b[2];
    if (g == 0 && vu) {
        out[pu * 3 + 0] = u0 + pb0;
        out[pu * 3 + 1] = u1 + pb1;
        out[pu * 3 + 2] = u2 + pb2;
    }
    if (g == 0 && vv) {
        out[pv * 3 + 0] = w0 + pb0;
        out[pv * 3 + 1] = w1 + pb1;
        out[pv * 3 + 2] = w2 + pb2;
    }
}

extern "C" void kernel_launch(void* const* d_in, const int* in_sizes, int n_in,
                              void* d_out, int out_size, void* d_ws, size_t ws_size,
                              hipStream_t stream) {
    const float* x      = (const float*)d_in[0];
    const float* lift_w = (const float*)d_in[1];
    const float* lift_b = (const float*)d_in[2];
    const float* W1     = (const float*)d_in[3];
    const float* b1     = (const float*)d_in[4];
    const float* W2     = (const float*)d_in[5];
    const float* b2     = (const float*)d_in[6];
    const float* dts    = (const float*)d_in[7];
    const float* proj_w = (const float*)d_in[8];
    const float* proj_b = (const float*)d_in[9];
    float* out = (float*)d_out;

    const int B = in_sizes[0] / DIN;

    prep_kernel<<<(6400 + 3200 + 64 + 255) / 256, 256, 0, stream>>>(
        lift_w, lift_b, W1, b1, W2, b2, proj_w, (char*)d_ws);

    const int tiles  = (B + 31) / 32;        // 32 points per wave
    const int blocks = (tiles + 7) / 8;      // 8 waves per block
    mlp_kernel<<<blocks, 256, 0, stream>>>(x, dts, proj_b,
                                           (const char*)d_ws, out, B);
}
